// ResBlock_64845416235416
// MI455X (gfx1250) — 16-bit and 8-bit matrix operand order repaired and checked by reading, compiled, not run on hardware
//
#include <hip/hip_runtime.h>
#include <hip/hip_bf16.h>


#define L_DIM  2048
#define C_DIM  512
#define BL_DIM 32768
#define MT     64
#define NTW    128

#define FP8_P1 ((unsigned char)0x38)
#define FP8_M1 ((unsigned char)0xB8)

typedef __attribute__((ext_vector_type(16))) _Float16 v16h;
typedef __attribute__((ext_vector_type(8)))  float    v8f;
typedef __attribute__((ext_vector_type(16))) int      v16i;

union FragH   { uint4 q[2]; v16h v; };
union FragF8A { uint2 d[8]; v16i v; };
union FragF8B { uint4 q[4]; v16i v; };

__global__ __launch_bounds__(256) void prep_sign_kernel(
    const float* __restrict__ x,  const float* __restrict__ w1,
    const float* __restrict__ w2, const float* __restrict__ wsc,
    unsigned char* __restrict__ sx, _Float16* __restrict__ w1sT,
    unsigned char* __restrict__ w2sT, unsigned char* __restrict__ wscT)
{
    long i = (long)blockIdx.x * blockDim.x + threadIdx.x;

    const long n4 = (long)BL_DIM * C_DIM / 4;
    if (i < n4) {
        float4 v = ((const float4*)x)[i];
        unsigned char* p = sx + i * 4;
        p[0] = (v.x >= 0.f) ? FP8_P1 : FP8_M1;
        p[1] = (v.y >= 0.f) ? FP8_P1 : FP8_M1;
        p[2] = (v.z >= 0.f) ? FP8_P1 : FP8_M1;
        p[3] = (v.w >= 0.f) ? FP8_P1 : FP8_M1;
    }
    const long NW = 3L * C_DIM * C_DIM;
    if (i < NW) {
        long t   = i / (C_DIM * C_DIM);
        long rem = i - t * (C_DIM * C_DIM);
        long co  = rem / C_DIM;
        long ci  = rem - co * C_DIM;
        long src = t * C_DIM * C_DIM + ci * C_DIM + co;
        w1sT[i] = (w1[src] >= 0.f) ? (_Float16)1.0f : (_Float16)-1.0f;
        w2sT[i] = (w2[src] >= 0.f) ? FP8_P1 : FP8_M1;
    }
    if (i < (long)C_DIM * C_DIM) {
        long co = i / C_DIM, ci = i - co * C_DIM;
        wscT[i] = (wsc[ci * C_DIM + co] >= 0.f) ? FP8_P1 : FP8_M1;
    }
}

__global__ __launch_bounds__(256) void conv1_wmma_f16_kernel(
    const float*    __restrict__ x,
    const _Float16* __restrict__ w1sT,
    const float* __restrict__ beta, const float* __restrict__ mean,
    const float* __restrict__ var,
    unsigned char* __restrict__ qh)
{
    __shared__ __align__(16) _Float16 Ash[2][MT][72];
    __shared__ __align__(16) _Float16 Bsh[2][NTW][72];

    const int tid  = threadIdx.x;
    const int lane = tid & 31;
    const int wave = tid >> 5;
    const int wm = wave & 3, wn = wave >> 2;
    const int hi = lane >> 4, ln = lane & 15;
    const long rowBase = (long)blockIdx.x * MT;
    const int  n0 = blockIdx.y * NTW;
    const int  NIT = 24;

    v8f acc[4] = {};

    auto stage = [&](int iter, int b) {
        const int t     = iter >> 3;
        const int k0    = (iter & 7) * 64;
        const int shift = 4 * (2 - t);
        const _Float16* Bsrc = w1sT + (long)t * C_DIM * C_DIM;
        {
            const int r  = tid >> 2;
            const int cg = (tid & 3) * 16;
            const long gr = rowBase + r;
            const int  l  = (int)(gr & (L_DIM - 1));
            float4 f[4] = {};
            if (l >= shift) {
                const float4* p =
                    (const float4*)(x + (gr - shift) * C_DIM + k0 + cg);
                f[0] = p[0]; f[1] = p[1]; f[2] = p[2]; f[3] = p[3];
            }
            _Float16* d = &Ash[b][r][cg];
#pragma unroll
            for (int j = 0; j < 4; ++j) {
                d[4 * j + 0] = (_Float16)f[j].x;
                d[4 * j + 1] = (_Float16)f[j].y;
                d[4 * j + 2] = (_Float16)f[j].z;
                d[4 * j + 3] = (_Float16)f[j].w;
            }
        }
        {
            const int n   = tid >> 1;
            const int off = (tid & 1) * 32;
            const uint4* p =
                (const uint4*)(Bsrc + (long)(n0 + n) * C_DIM + k0 + off);
            uint4* d = (uint4*)&Bsh[b][n][off];
            d[0] = p[0]; d[1] = p[1];
            if (k0 + 64 < C_DIM)
                __builtin_prefetch(Bsrc + (long)(n0 + n) * C_DIM + k0 + 64,
                                   0, 3);
        }
    };

    auto compute = [&](int b) {
#pragma unroll
        for (int ks = 0; ks < 2; ++ks) {
            FragH fa;
            const _Float16* ar = &Ash[b][wm * 16 + ln][ks * 32];
            fa.q[0] = *(const uint4*)(ar + hi * 8);
            fa.q[1] = *(const uint4*)(ar + 16 + hi * 8);
#pragma unroll
            for (int nt = 0; nt < 4; ++nt) {
                FragH fb;
                const _Float16* br =
                    &Bsh[b][wn * 64 + nt * 16 + ln][ks * 32];
                fb.q[0] = *(const uint4*)(br + hi * 8);
                fb.q[1] = *(const uint4*)(br + 16 + hi * 8);
                acc[nt] = __builtin_amdgcn_wmma_f32_16x16x32_f16(
                    false, fa.v, false, fb.v, (short)0, acc[nt],
                    false, false);
            }
        }
    };

    stage(0, 0);
    __syncthreads();
    for (int iter = 0; iter < NIT; ++iter) {
        const int cur = iter & 1;
        if (iter + 1 < NIT) stage(iter + 1, cur ^ 1);
        compute(cur);
        __syncthreads();
    }

#pragma unroll
    for (int nt = 0; nt < 4; ++nt) {
        const int co = n0 + wn * 64 + nt * 16 + ln;
        const float s   = rsqrtf(var[co] + 1e-3f);
        const float off = beta[co] - mean[co] * s;
#pragma unroll
        for (int e = 0; e < 8; ++e) {
            const int  m  = wm * 16 + e + 8 * hi;
            const long gr = rowBase + m;
            float h = acc[nt][e] * s + off;
            h = fmaxf(h, 0.f);
            qh[gr * C_DIM + co] = (h >= 0.f) ? FP8_P1 : FP8_M1;
        }
    }
}

__global__ __launch_bounds__(256) void conv2_shortcut_wmma_fp8_kernel(
    const unsigned char* __restrict__ qh,
    const unsigned char* __restrict__ sx,
    const unsigned char* __restrict__ w2sT,
    const unsigned char* __restrict__ wscT,
    const float* __restrict__ beta, const float* __restrict__ mean,
    const float* __restrict__ var,
    float* __restrict__ out)
{
    __shared__ __align__(16) unsigned char Ash[2][MT][144];
    __shared__ __align__(16) unsigned char Bsh[2][NTW][144];

    const int tid  = threadIdx.x;
    const int lane = tid & 31;
    const int wave = tid >> 5;
    const int wm = wave & 3, wn = wave >> 2;
    const int hi = lane >> 4, ln = lane & 15;
    const long rowBase = (long)blockIdx.x * MT;
    const int  n0 = blockIdx.y * NTW;
    const int  NIT = 16;

    v8f acc2[4] = {};
    v8f accS[4] = {};

    auto stage = [&](int iter, int b) {
        const int k0 = (iter >> 2) * 128;
        const int p  = iter & 3;
        const int shift = (p == 0) ? 0 : 4 * (3 - p);
        const unsigned char* Asrc = (p == 0) ? sx : qh;
        const unsigned char* Bsrc =
            (p == 0) ? wscT : (w2sT + (long)(p - 1) * C_DIM * C_DIM);
        {
            const int r  = tid >> 2;
            const int cg = (tid & 3) * 32;
            const long gr = rowBase + r;
            const int  l  = (int)(gr & (L_DIM - 1));
            uint4 v0 = {}, v1 = {};
            if (l >= shift) {
                const uint4* p4 =
                    (const uint4*)(Asrc + (gr - shift) * C_DIM + k0 + cg);
                v0 = p4[0]; v1 = p4[1];
            }
            uint4* d = (uint4*)&Ash[b][r][cg];
            d[0] = v0; d[1] = v1;
        }
        {
            const int n   = tid >> 1;
            const int off = (tid & 1) * 64;
            const uint4* p4 =
                (const uint4*)(Bsrc + (long)(n0 + n) * C_DIM + k0 + off);
            uint4* d = (uint4*)&Bsh[b][n][off];
            d[0] = p4[0]; d[1] = p4[1]; d[2] = p4[2]; d[3] = p4[3];
            if (k0 + 128 < C_DIM)
                __builtin_prefetch(Bsrc + (long)(n0 + n) * C_DIM + k0 + 128,
                                   0, 3);
        }
    };

    auto compute = [&](int iter, int b) {
        const int p = iter & 3;
        FragF8A fa;
        const unsigned char* ar = &Ash[b][wm * 16 + ln][0];
#pragma unroll
        for (int h2 = 0; h2 < 2; ++h2)
#pragma unroll
            for (int sgm = 0; sgm < 4; ++sgm)
                fa.d[h2 * 4 + sgm] =
                    *(const uint2*)(ar + h2 * 64 + (sgm >> 1) * 32 + hi * 16 + (sgm & 1) * 8);
#pragma unroll
        for (int nt = 0; nt < 4; ++nt) {
            FragF8B fb;
            const unsigned char* br = &Bsh[b][wn * 64 + nt * 16 + ln][0];
            fb.q[0] = *(const uint4*)(br + hi * 16);
            fb.q[1] = *(const uint4*)(br + 32 + hi * 16);
            fb.q[2] = *(const uint4*)(br + 64 + hi * 16);
            fb.q[3] = *(const uint4*)(br + 96 + hi * 16);
            if (p == 0)
                accS[nt] = __builtin_amdgcn_wmma_f32_16x16x128_fp8_fp8(
                    fa.v, fb.v, (short)0, accS[nt], false, false);
            else
                acc2[nt] = __builtin_amdgcn_wmma_f32_16x16x128_fp8_fp8(
                    fa.v, fb.v, (short)0, acc2[nt], false, false);
        }
    };

    stage(0, 0);
    __syncthreads();
    for (int iter = 0; iter < NIT; ++iter) {
        const int cur = iter & 1;
        if (iter + 1 < NIT) stage(iter + 1, cur ^ 1);
        compute(iter, cur);
        __syncthreads();
    }

#pragma unroll
    for (int nt = 0; nt < 4; ++nt) {
        const int co = n0 + wn * 64 + nt * 16 + ln;
        const float s   = rsqrtf(var[co] + 1e-3f);
        const float off = beta[co] - mean[co] * s;
#pragma unroll
        for (int e = 0; e < 8; ++e) {
            const int  m  = wm * 16 + e + 8 * hi;
            const long gr = rowBase + m;
            float h = acc2[nt][e] * s + off;
            h = fmaxf(h, 0.f);
            out[gr * C_DIM + co] = fmaxf(h + accS[nt][e], 0.f);
        }
    }
}

extern "C" void kernel_launch(void* const* d_in, const int* in_sizes, int n_in,
                              void* d_out, int out_size, void* d_ws, size_t ws_size,
                              hipStream_t stream) {
    const float* x    = (const float*)d_in[0];
    const float* w1   = (const float*)d_in[1];
    const float* w2   = (const float*)d_in[2];
    const float* wsc  = (const float*)d_in[3];
    const float* bn1b = (const float*)d_in[4];
    const float* bn1m = (const float*)d_in[5];
    const float* bn1v = (const float*)d_in[6];
    const float* bn2b = (const float*)d_in[7];
    const float* bn2m = (const float*)d_in[8];
    const float* bn2v = (const float*)d_in[9];
    float* out = (float*)d_out;

    char* ws = (char*)d_ws;
    unsigned char* sx   = (unsigned char*)(ws + 0);
    unsigned char* qh   = (unsigned char*)(ws + 16777216);
    _Float16*      w1sT = (_Float16*)     (ws + 33554432);
    unsigned char* w2sT = (unsigned char*)(ws + 35127296);
    unsigned char* wscT = (unsigned char*)(ws + 35913728);

    {
        const long n4 = (long)BL_DIM * C_DIM / 4;
        dim3 grid((unsigned)((n4 + 255) / 256));
        prep_sign_kernel<<<grid, 256, 0, stream>>>(x, w1, w2, wsc,
                                                   sx, w1sT, w2sT, wscT);
    }
    {
        dim3 grid(BL_DIM / MT, C_DIM / NTW);
        conv1_wmma_f16_kernel<<<grid, 256, 0, stream>>>(
            x, w1sT, bn1b, bn1m, bn1v, qh);
    }
    {
        dim3 grid(BL_DIM / MT, C_DIM / NTW);
        conv2_shortcut_wmma_fp8_kernel<<<grid, 256, 0, stream>>>(
            qh, sx, w2sT, wscT, bn2b, bn2m, bn2v, out);
    }
}
